// TransformerSelfAttention_73186242724055
// MI455X (gfx1250) — matrix operand order repaired and checked by reading, not compiled, not run on hardware
//
#include <hip/hip_runtime.h>
#include <cstdint>
#include <cstddef>

// Problem constants (match reference)
#define BB   2
#define SS   2048
#define HH   2048
#define NHH  16
#define HDD  128
#define MM   (BB * SS)              // 4096 rows in the flattened GEMMs
#define ATTN_SCALE 0.08838834764831845f  // 1/sqrt(128)

typedef __attribute__((ext_vector_type(16))) __bf16 v16bf;
typedef __attribute__((ext_vector_type(8)))  float  v8f;
typedef __attribute__((ext_vector_type(4)))  unsigned int u32x4;
typedef __attribute__((ext_vector_type(8)))  int i32x8;
typedef __attribute__((ext_vector_type(4)))  int i32x4;

union Frag { v16bf v; uint4 q[2]; };   // 32 bytes: one WMMA A/B operand (16 bf16/lane)
union Acc  { v8f  v;  float f[8]; };   // one WMMA C/D tile (8 f32/lane)

__device__ __forceinline__ v8f wmma_bf16(const Frag& a, const Frag& b, v8f c) {
  // D = A(16x32 bf16) * B(32x16 bf16) + C(16x16 f32)
  return __builtin_amdgcn_wmma_f32_16x16x32_bf16(
      /*neg_a=*/false, a.v, /*neg_b=*/false, b.v,
      /*c_mod=*/(short)0, c, /*reuse_a=*/false, /*reuse_b=*/false);
}

// Async global->LDS copy of 16 bytes per lane (GVS mode), tracked by ASYNCcnt.
__device__ __forceinline__ void async_copy_b128(unsigned lds_byte_addr,
                                                unsigned gbl_byte_off,
                                                unsigned long long gbl_base) {
  asm volatile("global_load_async_to_lds_b128 %0, %1, %2"
               :: "v"(lds_byte_addr), "v"(gbl_byte_off), "s"(gbl_base)
               : "memory");
}
__device__ __forceinline__ void async_wait0() {
  asm volatile("s_wait_asynccnt 0x0" ::: "memory");
}

// Tensor Data Mover: 2D tile load (bf16 elements) with optional LDS row
// padding. Descriptor packed per CDNA5 ISA section 8.3/8.4 (D# groups 0/1).
__device__ __forceinline__ void tdm_load_2d_bf16(
    unsigned lds_addr, const void* gaddr,
    unsigned tensor_d0, unsigned tensor_d1, unsigned d0_stride,
    unsigned tile_d0, unsigned tile_d1,
    unsigned pad_interval_code, unsigned pad_amount_code)
{
  unsigned long long ga = (unsigned long long)(uintptr_t)gaddr;
  u32x4 g0;
  g0[0] = 1u;                                    // count=1, user descriptor
  g0[1] = lds_addr;                              // lds_addr       [63:32]
  g0[2] = (unsigned)(ga & 0xffffffffu);          // global_addr lo [95:64]
  g0[3] = (unsigned)((ga >> 32) & 0x01ffffffu)   // global_addr hi [120:96]
        | (2u << 30);                            // type=2 (image) [127:126]
  i32x8 g1;
  g1[0] = (int)((1u << 16)                       // data_size = 2 bytes
        | (1u << 20)                             // pad_enable
        | (pad_interval_code << 22)
        | (pad_amount_code << 25));
  g1[1] = (int)((tensor_d0 & 0xffffu) << 16);    // tensor_dim0 lo  [79:48]
  g1[2] = (int)(((tensor_d0 >> 16) & 0xffffu)    // tensor_dim0 hi
        | ((tensor_d1 & 0xffffu) << 16));        // tensor_dim1 lo  [111:80]
  g1[3] = (int)(((tensor_d1 >> 16) & 0xffffu)
        | ((tile_d0 & 0xffffu) << 16));          // tile_dim0      [127:112]
  g1[4] = (int)(tile_d1 & 0xffffu);              // tile_dim1, tile_dim2=0
  g1[5] = (int)d0_stride;                        // tensor_dim0_stride lo32
  g1[6] = 0;                                     // stride hi, dim1_stride lo
  g1[7] = 0;
  i32x4 g2 = {0, 0, 0, 0};
  i32x4 g3 = {0, 0, 0, 0};
#if defined(__clang_major__) && __clang_major__ >= 23
  i32x8 g4 = {0, 0, 0, 0, 0, 0, 0, 0};
  __builtin_amdgcn_tensor_load_to_lds(g0, g1, g2, g3, g4, 0);
#else
  __builtin_amdgcn_tensor_load_to_lds(g0, g1, g2, g3, 0);
#endif
}

// ---------------------------------------------------------------------------
// Kernel 1: fused QKV projection.  NT-GEMM: out[m,n] = sum_k x[m,k]*W[n,k]+b[n]
// blockIdx.z selects {Wq,Wk,Wv}. Q,K stored as bf16 [B,S,H]; V stored
// TRANSPOSED per head as Vt[B,NH,HD,S] so that P@V later is also an NT-GEMM.
// ---------------------------------------------------------------------------
__global__ __launch_bounds__(256) void k_qkv(
    const float* __restrict__ x,
    const float* __restrict__ Wq, const float* __restrict__ Wk,
    const float* __restrict__ Wv,
    const float* __restrict__ bq, const float* __restrict__ bk,
    const float* __restrict__ bv,
    __bf16* __restrict__ qbuf, __bf16* __restrict__ kbuf,
    __bf16* __restrict__ vtbuf)
{
  __shared__ __align__(16) __bf16 At[128][72];   // 128 rows x 64 K (bf16, padded)
  __shared__ __align__(16) __bf16 Bt[128][72];

  const int n0  = blockIdx.x * 128;
  const int m0  = blockIdx.y * 128;
  const int mat = blockIdx.z;
  const float* W    = (mat == 0) ? Wq : (mat == 1) ? Wk : Wv;
  const float* bias = (mat == 0) ? bq : (mat == 1) ? bk : bv;

  const int tid  = threadIdx.x;
  const int wave = tid >> 5, lane = tid & 31;
  const int l16  = lane & 15, h = lane >> 4;
  const int mr   = wave & 3;   // 4 row-groups of 32
  const int nc   = wave >> 2;  // 2 col-groups of 64

  Acc acc[2][4];
  #pragma unroll
  for (int t = 0; t < 2; ++t)
    #pragma unroll
    for (int u = 0; u < 4; ++u)
      #pragma unroll
      for (int v = 0; v < 8; ++v) acc[t][u].f[v] = 0.f;

  for (int k0 = 0; k0 < HH; k0 += 64) {
    // Stage A (x) and B (W) tiles: batch all global loads first so they
    // pipeline, then convert f32 -> bf16 and store to LDS.
    float4 ar[8], br[8];
    #pragma unroll
    for (int t = 0; t < 8; ++t) {
      int i = tid + t * 256;               // 2048 float4 items per matrix
      int row = i >> 4, c4 = i & 15;
      ar[t] = *reinterpret_cast<const float4*>(
          x + (size_t)(m0 + row) * HH + k0 + c4 * 4);
      br[t] = *reinterpret_cast<const float4*>(
          W + (size_t)(n0 + row) * HH + k0 + c4 * 4);
    }
    #pragma unroll
    for (int t = 0; t < 8; ++t) {
      int i = tid + t * 256;
      int row = i >> 4, c4 = i & 15;
      At[row][c4 * 4 + 0] = (__bf16)ar[t].x; At[row][c4 * 4 + 1] = (__bf16)ar[t].y;
      At[row][c4 * 4 + 2] = (__bf16)ar[t].z; At[row][c4 * 4 + 3] = (__bf16)ar[t].w;
      Bt[row][c4 * 4 + 0] = (__bf16)br[t].x; Bt[row][c4 * 4 + 1] = (__bf16)br[t].y;
      Bt[row][c4 * 4 + 2] = (__bf16)br[t].z; Bt[row][c4 * 4 + 3] = (__bf16)br[t].w;
    }
    if (k0 + 64 < HH) {  // hint next chunk into cache (global_prefetch_b8)
      __builtin_prefetch(x + (size_t)(m0 + (tid >> 1)) * HH + (k0 + 64) + (tid & 1) * 32, 0, 1);
      __builtin_prefetch(W + (size_t)(n0 + (tid >> 1)) * HH + (k0 + 64) + (tid & 1) * 32, 0, 1);
    }
    __syncthreads();

    #pragma unroll
    for (int kk = 0; kk < 64; kk += 32) {
      Frag a[2], b[4];
      #pragma unroll
      for (int t = 0; t < 2; ++t) {   // A-frag: lane=row, K chunks at 8h / 16+8h
        const __bf16* p = &At[mr * 32 + t * 16 + l16][kk + 8 * h];
        a[t].q[0] = *reinterpret_cast<const uint4*>(p);
        a[t].q[1] = *reinterpret_cast<const uint4*>(p + 16);
      }
      #pragma unroll
      for (int u = 0; u < 4; ++u) {   // B-frag: lane=n, K chunks at 8h / 16+8h
        const __bf16* p = &Bt[nc * 64 + u * 16 + l16][kk + 8 * h];
        b[u].q[0] = *reinterpret_cast<const uint4*>(p);
        b[u].q[1] = *reinterpret_cast<const uint4*>(p + 16);
      }
      #pragma unroll
      for (int t = 0; t < 2; ++t)
        #pragma unroll
        for (int u = 0; u < 4; ++u)
          acc[t][u].v = wmma_bf16(a[t], b[u], acc[t][u].v);
    }
    __syncthreads();
  }

  // Epilogue: bias add, bf16 convert, store (V goes out transposed per head).
  #pragma unroll
  for (int t = 0; t < 2; ++t)
    #pragma unroll
    for (int u = 0; u < 4; ++u) {
      int n = n0 + nc * 64 + u * 16 + l16;
      float bvn = bias[n];
      #pragma unroll
      for (int v = 0; v < 8; ++v) {
        int m = m0 + mr * 32 + t * 16 + v + 8 * h;
        __bf16 o = (__bf16)(acc[t][u].f[v] + bvn);
        if (mat == 0) {
          qbuf[(size_t)m * HH + n] = o;
        } else if (mat == 1) {
          kbuf[(size_t)m * HH + n] = o;
        } else {
          int bi = m >> 11, qi = m & (SS - 1);
          int hi = n >> 7,  di = n & (HDD - 1);
          vtbuf[(((size_t)bi * NHH + hi) * HDD + di) * SS + qi] = o;
        }
      }
    }
}

// ---------------------------------------------------------------------------
// Kernel 2: flash-style attention. One block = (q-tile of 128, head, batch);
// 8 waves, each wave owns 16 query rows. K tile staged by the Tensor Data
// Mover (padded directly into the LDS layout); V^T tile staged with async
// global->LDS loads. Online softmax with 16-lane shuffle reductions; P
// round-trips through per-wave LDS to convert C-layout -> A-layout.
// ---------------------------------------------------------------------------
__global__ __launch_bounds__(256) void k_attn(
    const __bf16* __restrict__ qbuf, const __bf16* __restrict__ kbuf,
    const __bf16* __restrict__ vtbuf, const float* __restrict__ mask,
    __bf16* __restrict__ ctx)
{
  __shared__ __align__(16) __bf16 Kt[64][136];   // 64 keys x 128 dims (+pad)
  __shared__ __align__(16) __bf16 Vt[128][72];   // 128 dims x 64 keys (V^T)
  __shared__ __align__(16) __bf16 Pb[8][16][40]; // per-wave P staging

  const int qb = blockIdx.x, hy = blockIdx.y, bz = blockIdx.z;
  const int tid = threadIdx.x, wave = tid >> 5, lane = tid & 31;
  const int l16 = lane & 15, h = lane >> 4;
  const int qrow0 = qb * 128 + wave * 16;

  const unsigned KtBase = (unsigned)(uintptr_t)&Kt[0][0];
  const unsigned VtBase = (unsigned)(uintptr_t)&Vt[0][0];
  const unsigned long long vtGbl = (unsigned long long)(uintptr_t)(
      vtbuf + ((size_t)bz * NHH + hy) * HDD * SS);

  // Preload Q A-frags for the whole head dim (4 x K=32).
  Frag qf[4];
  const __bf16* qrp =
      qbuf + (size_t)(bz * SS + qrow0 + l16) * HH + hy * HDD;
  #pragma unroll
  for (int kk4 = 0; kk4 < 4; ++kk4) {
    qf[kk4].q[0] = *reinterpret_cast<const uint4*>(qrp + kk4 * 32 + 8 * h);
    qf[kk4].q[1] = *reinterpret_cast<const uint4*>(qrp + kk4 * 32 + 16 + 8 * h);
  }

  Acc o[8];                       // 16 q-rows x 128 dims accumulator
  #pragma unroll
  for (int d = 0; d < 8; ++d)
    #pragma unroll
    for (int v = 0; v < 8; ++v) o[d].f[v] = 0.f;
  float mrow[8], lrow[8];
  #pragma unroll
  for (int v = 0; v < 8; ++v) { mrow[v] = -1e30f; lrow[v] = 0.f; }

  for (int j0 = 0; j0 < SS; j0 += 64) {
    // --- K tile via Tensor Data Mover: 64 rows x 128 bf16, row stride H,
    //     LDS-padded 16B per 256B row -> lands exactly in Kt[64][136]. ---
    if (wave == 0) {
      tdm_load_2d_bf16(KtBase,
                       kbuf + ((size_t)(bz * SS + j0)) * HH + hy * HDD,
                       /*tensor_d0=*/HDD, /*tensor_d1=*/SS, /*d0_stride=*/HH,
                       /*tile_d0=*/HDD, /*tile_d1=*/64,
                       /*pad_interval(64 dw)=*/5u, /*pad_amount(4 dw)=*/3u);
    }
    // --- V^T tile via async global->LDS (1024 x 16B, no VGPR data path) ---
    #pragma unroll
    for (int t = 0; t < 4; ++t) {
      int i = tid + t * 256;
      int vr = i >> 3, vc = i & 7;
      async_copy_b128(VtBase + (unsigned)(vr * 72 + vc * 8) * 2u,
                      (unsigned)((vr * SS + j0 + vc * 8) * 2),
                      vtGbl);
    }
    async_wait0();
    if (wave == 0) __builtin_amdgcn_s_wait_tensorcnt(0);
    __syncthreads();

    #pragma unroll
    for (int jj = 0; jj < 64; jj += 32) {
      // ---- S = Q K^T for a 16x32 score strip (two 16x16 tiles) ----
      Acc s[2];
      #pragma unroll
      for (int v = 0; v < 8; ++v) { s[0].f[v] = 0.f; s[1].f[v] = 0.f; }
      #pragma unroll
      for (int kk4 = 0; kk4 < 4; ++kk4)
        #pragma unroll
        for (int u = 0; u < 2; ++u) {
          Frag b;
          const __bf16* p = &Kt[jj + u * 16 + l16][kk4 * 32 + 8 * h];
          b.q[0] = *reinterpret_cast<const uint4*>(p);
          b.q[1] = *reinterpret_cast<const uint4*>(p + 16);
          s[u].v = wmma_bf16(qf[kk4], b, s[u].v);
        }

      // ---- scale + additive mask (mask is per-key, [B,1,1,S]) ----
      float mk0 = mask[(size_t)bz * SS + j0 + jj + l16];
      float mk1 = mask[(size_t)bz * SS + j0 + jj + 16 + l16];
      float cm[8];
      #pragma unroll
      for (int v = 0; v < 8; ++v) {
        s[0].f[v] = s[0].f[v] * ATTN_SCALE + mk0;
        s[1].f[v] = s[1].f[v] * ATTN_SCALE + mk1;
        cm[v] = fmaxf(s[0].f[v], s[1].f[v]);
      }
      // row max across the 16 lanes of each half (rows live in VGPR index)
      #pragma unroll
      for (int off = 1; off < 16; off <<= 1)
        #pragma unroll
        for (int v = 0; v < 8; ++v)
          cm[v] = fmaxf(cm[v], __shfl_xor(cm[v], off));

      float alpha[8], rs[8];
      #pragma unroll
      for (int v = 0; v < 8; ++v) {
        float mn = fmaxf(mrow[v], cm[v]);
        alpha[v] = __expf(mrow[v] - mn);
        mrow[v]  = mn;
        s[0].f[v] = __expf(s[0].f[v] - mn);
        s[1].f[v] = __expf(s[1].f[v] - mn);
        rs[v] = s[0].f[v] + s[1].f[v];
      }
      #pragma unroll
      for (int off = 1; off < 16; off <<= 1)
        #pragma unroll
        for (int v = 0; v < 8; ++v)
          rs[v] += __shfl_xor(rs[v], off);
      #pragma unroll
      for (int v = 0; v < 8; ++v) {
        lrow[v] = lrow[v] * alpha[v] + rs[v];
        #pragma unroll
        for (int d = 0; d < 8; ++d) o[d].f[v] *= alpha[v];
      }

      // ---- P: C-layout -> A-layout via per-wave LDS (DS ops in-order) ----
      #pragma unroll
      for (int v = 0; v < 8; ++v) {
        Pb[wave][v + 8 * h][l16]      = (__bf16)s[0].f[v];
        Pb[wave][v + 8 * h][16 + l16] = (__bf16)s[1].f[v];
      }
      Frag pf;
      const __bf16* pp = &Pb[wave][l16][0];
      pf.q[0] = *reinterpret_cast<const uint4*>(pp + 8 * h);
      pf.q[1] = *reinterpret_cast<const uint4*>(pp + 16 + 8 * h);

      // ---- O += P @ V  (NT via V^T tile) ----
      #pragma unroll
      for (int d = 0; d < 8; ++d) {
        Frag b;
        const __bf16* vp = &Vt[d * 16 + l16][jj + 8 * h];
        b.q[0] = *reinterpret_cast<const uint4*>(vp);
        b.q[1] = *reinterpret_cast<const uint4*>(vp + 16);
        o[d].v = wmma_bf16(pf, b, o[d].v);
      }
    }
    __syncthreads();
  }

  // Normalize by softmax denominator and store ctx (bf16, [B,S,H] layout).
  #pragma unroll
  for (int v = 0; v < 8; ++v) {
    float inv = 1.f / lrow[v];
    size_t rowoff = (size_t)(bz * SS + qrow0 + v + 8 * h) * HH + hy * HDD;
    #pragma unroll
    for (int d = 0; d < 8; ++d)
      ctx[rowoff + d * 16 + l16] = (__bf16)(o[d].f[v] * inv);
  }
}

// ---------------------------------------------------------------------------
// Kernel 3: output projection. A = ctx (bf16, async-copied to LDS),
// B = Wo (f32 -> bf16), f32 out.
// ---------------------------------------------------------------------------
__global__ __launch_bounds__(256) void k_oproj(
    const __bf16* __restrict__ ctx, const float* __restrict__ Wo,
    const float* __restrict__ bo, float* __restrict__ out)
{
  __shared__ __align__(16) __bf16 At[128][72];
  __shared__ __align__(16) __bf16 Bt[128][72];

  const int n0 = blockIdx.x * 128;
  const int m0 = blockIdx.y * 128;
  const int tid  = threadIdx.x;
  const int wave = tid >> 5, lane = tid & 31;
  const int l16  = lane & 15, h = lane >> 4;
  const int mr   = wave & 3, nc = wave >> 2;

  const unsigned AtBase = (unsigned)(uintptr_t)&At[0][0];

  Acc acc[2][4];
  #pragma unroll
  for (int t = 0; t < 2; ++t)
    #pragma unroll
    for (int u = 0; u < 4; ++u)
      #pragma unroll
      for (int v = 0; v < 8; ++v) acc[t][u].f[v] = 0.f;

  for (int k0 = 0; k0 < HH; k0 += 64) {
    // A tile (already bf16): async global->LDS copy, 1024 x 16B.
    const unsigned long long aGbl =
        (unsigned long long)(uintptr_t)(ctx + (size_t)m0 * HH + k0);
    #pragma unroll
    for (int t = 0; t < 4; ++t) {
      int i = tid + t * 256;
      int row = i >> 3, c8 = i & 7;
      async_copy_b128(AtBase + (unsigned)(row * 72 + c8 * 8) * 2u,
                      (unsigned)((row * HH + c8 * 8) * 2),
                      aGbl);
    }
    // B tile: batched f32 loads, convert -> bf16.
    float4 br[8];
    #pragma unroll
    for (int t = 0; t < 8; ++t) {
      int i = tid + t * 256;
      int row = i >> 4, c4 = i & 15;
      br[t] = *reinterpret_cast<const float4*>(
          Wo + (size_t)(n0 + row) * HH + k0 + c4 * 4);
    }
    #pragma unroll
    for (int t = 0; t < 8; ++t) {
      int i = tid + t * 256;
      int row = i >> 4, c4 = i & 15;
      Bt[row][c4 * 4 + 0] = (__bf16)br[t].x; Bt[row][c4 * 4 + 1] = (__bf16)br[t].y;
      Bt[row][c4 * 4 + 2] = (__bf16)br[t].z; Bt[row][c4 * 4 + 3] = (__bf16)br[t].w;
    }
    async_wait0();
    __syncthreads();

    #pragma unroll
    for (int kk = 0; kk < 64; kk += 32) {
      Frag a[2], b[4];
      #pragma unroll
      for (int t = 0; t < 2; ++t) {
        const __bf16* p = &At[mr * 32 + t * 16 + l16][kk + 8 * h];
        a[t].q[0] = *reinterpret_cast<const uint4*>(p);
        a[t].q[1] = *reinterpret_cast<const uint4*>(p + 16);
      }
      #pragma unroll
      for (int u = 0; u < 4; ++u) {
        const __bf16* p = &Bt[nc * 64 + u * 16 + l16][kk + 8 * h];
        b[u].q[0] = *reinterpret_cast<const uint4*>(p);
        b[u].q[1] = *reinterpret_cast<const uint4*>(p + 16);
      }
      #pragma unroll
      for (int t = 0; t < 2; ++t)
        #pragma unroll
        for (int u = 0; u < 4; ++u)
          acc[t][u].v = wmma_bf16(a[t], b[u], acc[t][u].v);
    }
    __syncthreads();
  }

  #pragma unroll
  for (int t = 0; t < 2; ++t)
    #pragma unroll
    for (int u = 0; u < 4; ++u) {
      int n = n0 + nc * 64 + u * 16 + l16;
      float bn = bo[n];
      #pragma unroll
      for (int v = 0; v < 8; ++v) {
        int m = m0 + mr * 32 + t * 16 + v + 8 * h;
        out[(size_t)m * HH + n] = acc[t][u].f[v] + bn;
      }
    }
}

// ---------------------------------------------------------------------------
extern "C" void kernel_launch(void* const* d_in, const int* in_sizes, int n_in,
                              void* d_out, int out_size, void* d_ws,
                              size_t ws_size, hipStream_t stream) {
  (void)in_sizes; (void)n_in; (void)out_size; (void)ws_size;
  const float* x    = (const float*)d_in[0];
  const float* mask = (const float*)d_in[1];
  const float* Wq   = (const float*)d_in[2];
  const float* bq   = (const float*)d_in[3];
  const float* Wk   = (const float*)d_in[4];
  const float* bk   = (const float*)d_in[5];
  const float* Wv   = (const float*)d_in[6];
  const float* bv   = (const float*)d_in[7];
  const float* Wo   = (const float*)d_in[8];
  const float* bo   = (const float*)d_in[9];
  float* out = (float*)d_out;

  char* ws = (char*)d_ws;
  const size_t SZ = (size_t)MM * HH * 2;   // 16 MiB per bf16 activation buffer
  __bf16* qbuf  = (__bf16*)(ws);
  __bf16* kbuf  = (__bf16*)(ws + SZ);
  __bf16* vtbuf = (__bf16*)(ws + 2 * SZ);
  __bf16* ctx   = (__bf16*)(ws + 3 * SZ);  // total 64 MiB workspace

  k_qkv  <<<dim3(HH / 128, MM / 128, 3), 256, 0, stream>>>(
      x, Wq, Wk, Wv, bq, bk, bv, qbuf, kbuf, vtbuf);
  k_attn <<<dim3(SS / 128, NHH, BB), 256, 0, stream>>>(
      qbuf, kbuf, vtbuf, mask, ctx);
  k_oproj<<<dim3(HH / 128, MM / 128), 256, 0, stream>>>(ctx, Wo, bo, out);
}
